// BasicClassifier_59072980189355
// MI455X (gfx1250) — compile-verified, matrix operand order checked by reading, not run on hardware
//
#include <hip/hip_runtime.h>
#include <hip/hip_bf16.h>

typedef __attribute__((ext_vector_type(16))) _Float16 v16h;
typedef __attribute__((ext_vector_type(8)))  float    v8f;

#define T_STEPS 1000
#define LIN     30
#define LO      24
#define CHANNELS 16
#define KDIM    384      // 16*24
#define OUTC    35
#define NTILES  3        // ceil(35/16)
#define KTILES  12       // 384/32
#define BTILE   16       // batch per block

// ---------------------------------------------------------------------------
// Prep: convert fc_w (35x384 f32, row-major) into f16 B-fragments, fragment-
// ordered so each WMMA lane loads its 16 halves (32B) contiguously.
// B is KxN (384x48 padded). 16-bit fragment layout (mirror of A layout):
//   lane = g*16 + n,  per-lane halves: K = base(r) + g*8 + h, with
//   base(r) = r<4 ? 2r : 16+2(r-4).
// wsB flat index = ((kt*3 + nt)*32 + lane)*16 + r*2 + h
// ---------------------------------------------------------------------------
__global__ __launch_bounds__(256) void prep_fcw(const float* __restrict__ fc_w,
                                                _Float16* __restrict__ wsB) {
  int idx = blockIdx.x * 256 + threadIdx.x;        // 0 .. 12*3*32*16-1
  if (idx >= KTILES * NTILES * 32 * 16) return;
  int h16  = idx & 15;
  int lane = (idx >> 4) & 31;
  int ktnt = idx >> 9;
  int nt   = ktnt % NTILES;
  int kt   = ktnt / NTILES;
  int r    = h16 >> 1;
  int hh   = h16 & 1;
  int g    = lane >> 4;
  int n    = lane & 15;
  int base = (r < 4) ? (r * 2) : (16 + (r - 4) * 2);
  int K    = kt * 32 + base + g * 8 + hh;
  int N    = nt * 16 + n;
  float v  = (N < OUTC) ? fc_w[N * KDIM + K] : 0.0f;
  wsB[idx] = (_Float16)v;
}

// ---------------------------------------------------------------------------
// Fused SNN: conv1d + LIF1 + (spike GEMM via v_wmma_f32_16x16x32_f16) + LIF2
// + time-mean, all inside the T loop. One block per 16-batch tile.
// ---------------------------------------------------------------------------
__global__ __launch_bounds__(256) void snn_fused(
    const float* __restrict__ x,       // (256,1000,30)
    const float* __restrict__ conv_w,  // (16,1,7)
    const float* __restrict__ conv_b,  // (16)
    const float* __restrict__ fc_b,    // (35)
    const _Float16* __restrict__ wsB,  // B fragments
    float* __restrict__ out)           // (256,35)
{
  __shared__ __align__(32) _Float16 frag[KTILES * 32 * 16]; // spike A-fragments
  __shared__ float xs[BTILE * LIN];                          // x tile for step t

  const int tid  = threadIdx.x;
  const int lane = tid & 31;
  const int wave = tid >> 5;
  const int b    = tid >> 4;      // batch within tile (0..15)
  const int ch   = tid & 15;      // conv channel     (0..15)
  const int b0   = blockIdx.x * BTILE;

  // Per-thread conv weights + LIF1 state (registers)
  float w[7];
#pragma unroll
  for (int j = 0; j < 7; ++j) w[j] = conv_w[ch * 7 + j];
  const float cb = conv_b[ch];
  float mem1[LO];
#pragma unroll
  for (int p = 0; p < LO; ++p) mem1[p] = 0.0f;

  // WMMA waves: resident B fragments + fragment-register LIF2 state
  const int nt = wave;
  v16h Bfrag[KTILES];
  v8f mem2  = {0.f,0.f,0.f,0.f,0.f,0.f,0.f,0.f};
  v8f sumv  = {0.f,0.f,0.f,0.f,0.f,0.f,0.f,0.f};
  v8f cinit = {0.f,0.f,0.f,0.f,0.f,0.f,0.f,0.f};
  if (wave < NTILES) {
#pragma unroll
    for (int kt = 0; kt < KTILES; ++kt)
      Bfrag[kt] = *reinterpret_cast<const v16h*>(
          wsB + (size_t)((kt * NTILES + nt) * 32 + lane) * 16);
    int N = nt * 16 + (lane & 15);
    float fb = (N < OUTC) ? fc_b[N] : 0.0f;
#pragma unroll
    for (int i = 0; i < 8; ++i) cinit[i] = fb;
  }

  for (int t = 0; t < T_STEPS; ++t) {
    __syncthreads();  // frag/xs of step t-1 fully consumed
    // Cooperative x-tile load: 16 batches x 30 floats
    for (int i = tid; i < BTILE * LIN; i += 256) {
      int bb = i / LIN, pos = i - bb * LIN;
      const float* src = x + ((size_t)(b0 + bb) * T_STEPS + t) * LIN + pos;
      xs[i] = *src;
      if (t + 1 < T_STEPS) __builtin_prefetch(src + LIN, 0, 0); // next step
    }
    __syncthreads();

    // Conv1d (7 taps) + LIF1 + spike -> A-fragment in LDS
#pragma unroll
    for (int p = 0; p < LO; ++p) {
      float c1 = cb;
#pragma unroll
      for (int j = 0; j < 7; ++j) c1 = fmaf(w[j], xs[b * LIN + p + j], c1);
      float m  = mem1[p];
      float r1 = (m > 1.0f) ? 1.0f : 0.0f;         // reset-by-subtraction
      m = 0.9f * m + c1 - r1;
      mem1[p] = m;
      _Float16 s = (m > 1.0f) ? (_Float16)1.0f : (_Float16)0.0f;
      int k  = ch * LO + p;
      int kt = k >> 5, kk = k & 31;
      int hh = kk & 1;
      int g  = (kk >> 3) & 1;
      int r  = ((kk >> 4) & 1) * 4 + ((kk & 7) >> 1);
      frag[(size_t)((kt << 5) + (g << 4) + b) * 16 + r * 2 + hh] = s;
    }
    __syncthreads();

    // Spike GEMM (16x48x384 in f16) + LIF2, all in fragment registers
    if (wave < NTILES) {   // wave-uniform: EXEC all-1s for WMMA
      v8f acc = cinit;     // c2 starts at broadcast fc_b
#pragma unroll
      for (int kt = 0; kt < KTILES; ++kt) {
        v16h a = *reinterpret_cast<const v16h*>(frag + (size_t)((kt << 5) + lane) * 16);
        acc = __builtin_amdgcn_wmma_f32_16x16x32_f16(
            false, a, false, Bfrag[kt], (short)0, acc, false, false);
      }
#pragma unroll
      for (int i = 0; i < 8; ++i) {
        float m2 = mem2[i];
        float r2 = (m2 > 1.0f) ? 1.0f : 0.0f;
        m2 = 0.9f * m2 + acc[i] - r2;
        mem2[i] = m2;
        sumv[i] += m2;
      }
    }
  }

  // Write time-mean from C-fragment layout: N=lane&15, M=(lane>>4)*8 + i
  if (wave < NTILES) {
#pragma unroll
    for (int i = 0; i < 8; ++i) {
      int M = ((lane >> 4) << 3) + i;
      int N = nt * 16 + (lane & 15);
      if (N < OUTC)
        out[(size_t)(b0 + M) * OUTC + N] = sumv[i] * (1.0f / (float)T_STEPS);
    }
  }
}

extern "C" void kernel_launch(void* const* d_in, const int* in_sizes, int n_in,
                              void* d_out, int out_size, void* d_ws, size_t ws_size,
                              hipStream_t stream) {
  const float* x      = (const float*)d_in[0];
  const float* conv_w = (const float*)d_in[1];
  const float* conv_b = (const float*)d_in[2];
  const float* fc_w   = (const float*)d_in[3];
  const float* fc_b   = (const float*)d_in[4];
  float* out = (float*)d_out;
  _Float16* wsB = (_Float16*)d_ws;   // 12*3*32*16 halves = 36,864 bytes

  // Build f16 B fragments for the FC weights (fragment-ordered)
  prep_fcw<<<(KTILES * NTILES * 32 * 16 + 255) / 256, 256, 0, stream>>>(fc_w, wsB);
  // Fused SNN scan: one block per 16-batch tile
  snn_fused<<<256 / BTILE, 256, 0, stream>>>(x, conv_w, conv_b, fc_b, wsB, out);
}
